// RNNModel_78529182040511
// MI455X (gfx1250) — compile-verified, matrix operand order checked by reading, not run on hardware
//
#include <hip/hip_runtime.h>
#include <math.h>

// RNN LM forward for MI455X (gfx1250), wave32.
// T=128 B=32 V=50000 E=H=512 L=2.
//
// - Decoder (210 GFLOP, dominant GEMM): split-bf16 WMMA 16x16x32, 3-term
//   compensation (Ah*Bh + Ah*Bl + Al*Bh) -> ~2^-17 rel. error at ~8x the
//   FLOP/instruction of f32 WMMA. M-blocked 4x per wave so each B fragment
//   read from L2 feeds 12 WMMAs (L2 traffic 26 GB -> 6.5 GB). 819 MB output
//   via nontemporal stores so dec_W's bf16 planes (102 MB) stay L2-resident.
// - Recurrence is launch/sync-bound (serial chain). Layers 0 and 1 are
//   software-pipelined into ONE launch per time index (130 launches instead
//   of 256), with layer-1's input projection hoisted off the critical path
//   as a third pipeline stage writing a 2-slot ring buffer.
// - Recurrent math stays f32 WMMA 16x16x4 (error compounds over 128 steps).

typedef __attribute__((ext_vector_type(2)))  float  v2f;
typedef __attribute__((ext_vector_type(8)))  float  v8f;
typedef __attribute__((ext_vector_type(8)))  __bf16 v8bf;
typedef __attribute__((ext_vector_type(16))) __bf16 v16bf;

constexpr int T_ = 128, B_ = 32, V_ = 50000, E_ = 512, H_ = 512;
constexpr int KD    = 512;       // K dim of every GEMM here (E == H == 512)
constexpr int G4H   = 4 * H_;    // 2048 (gate width)
constexpr int TB    = T_ * B_;   // 4096
constexpr int LDSP  = KD + 4;    // f32 LDS pad: 516 % 64 banks == 4 -> conflict-free
constexpr int LDSPH = KD + 8;    // bf16 LDS pad: 520 halfs = 1040 B -> 4-bank skew, 16B aligned
constexpr int MB    = 4;         // decoder: M-tiles per block (64 rows, 4x B-reuse)

__device__ __forceinline__ float sigm(float x) { return 1.0f / (1.0f + __expf(-x)); }

// ---------------------------------------------------------------------------
// f32 WMMA GEMM, K fixed 512:
//   C[M x N] = A[M x 512] (rows optionally gathered via aidx) * Bw[N x 512]^T
//              + bias1[N] + bias2[N]
// ---------------------------------------------------------------------------
template <bool NT>
__global__ void gemm_wmma_k512(const float* __restrict__ A, const int* __restrict__ aidx,
                               const float* __restrict__ Bw,
                               const float* __restrict__ bias1, const float* __restrict__ bias2,
                               float* __restrict__ C, int N, int ldc) {
  __shared__ float As[16][LDSP];
  const int tid  = threadIdx.x;
  const int lane = tid & 31;
  const int wave = tid >> 5;
  const int m0   = blockIdx.y * 16;

  for (int i = tid; i < 16 * KD; i += 256) {
    int r = i >> 9;
    int c = i & (KD - 1);
    int arow = aidx ? aidx[m0 + r] : (m0 + r);
    As[r][c] = A[(size_t)arow * KD + c];
  }
  __syncthreads();

  const int ntile = blockIdx.x * 8 + wave;
  const int n0    = ntile * 16;
  if (n0 >= N) return;

  const int half = lane >> 4;
  const int l16  = lane & 15;
  const float* ap = &As[l16][half << 1];
  const float* bp = Bw + (size_t)(n0 + l16) * KD + (half << 1);

  v8f acc = {};
#pragma unroll 4
  for (int k = 0; k < KD; k += 4) {
    v2f af = *(const v2f*)(ap + k);
    v2f bf = *(const v2f*)(bp + k);
    acc = __builtin_amdgcn_wmma_f32_16x16x4_f32(false, af, false, bf,
                                                (short)0, acc, false, false);
  }

  const int n = n0 + l16;
  float bv = 0.0f;
  if (bias1) bv += bias1[n];
  if (bias2) bv += bias2[n];
#pragma unroll
  for (int r = 0; r < 8; ++r) {
    int m   = m0 + r + (half << 3);
    float v = acc[r] + bv;
    if (NT) __builtin_nontemporal_store(v, &C[(size_t)m * ldc + n]);
    else    C[(size_t)m * ldc + n] = v;
  }
}

// ---------------------------------------------------------------------------
// Split f32 -> bf16 hi/lo planes. hi = truncate(x), lo = bf16(x - hi) (exact
// residual in f32).
// ---------------------------------------------------------------------------
__global__ void split_bf16(const float* __restrict__ X, __bf16* __restrict__ Xh,
                           __bf16* __restrict__ Xl, unsigned n) {
  unsigned i = blockIdx.x * 256u + threadIdx.x;
  if (i >= n) return;
  float x = X[i];
  unsigned xb = __builtin_bit_cast(unsigned, x);
  unsigned hb = xb & 0xFFFF0000u;
  float lo = x - __builtin_bit_cast(float, hb);
  unsigned lb = __builtin_bit_cast(unsigned, lo);
  Xh[i] = __builtin_bit_cast(__bf16, (unsigned short)(hb >> 16));
  Xl[i] = __builtin_bit_cast(__bf16, (unsigned short)(lb >> 16));
}

// ---------------------------------------------------------------------------
// Decoder: C[4096 x 50000] = A[4096 x 512] * B[50000 x 512]^T + bias.
// A split to bf16 hi/lo during LDS staging (64 rows = MB M-tiles); B
// pre-split in global memory. Each wave: 1 N-tile x MB M-tiles -> each B
// fragment feeds 3*MB WMMAs; MB independent accumulator chains give ILP.
// 16-bit fragment layout: lane<16 holds K {0..7,16..23}, lane>=16 holds
// K {8..15,24..31} -> two b128 chunks per fragment.
// ---------------------------------------------------------------------------
__global__ void gemm_dec_bf16split(const float* __restrict__ A,
                                   const __bf16* __restrict__ Bh,
                                   const __bf16* __restrict__ Bl,
                                   const float* __restrict__ bias,
                                   float* __restrict__ C) {
  __shared__ __bf16 Ah[MB * 16][LDSPH];
  __shared__ __bf16 Al[MB * 16][LDSPH];
  const int tid  = threadIdx.x;
  const int lane = tid & 31;
  const int wave = tid >> 5;
  const int m0   = blockIdx.y * (MB * 16);

  for (int i = tid; i < MB * 16 * KD; i += 256) {
    int r = i >> 9;
    int c = i & (KD - 1);
    float x = A[(size_t)(m0 + r) * KD + c];
    unsigned xb = __builtin_bit_cast(unsigned, x);
    unsigned hb = xb & 0xFFFF0000u;
    float lo = x - __builtin_bit_cast(float, hb);
    unsigned lb = __builtin_bit_cast(unsigned, lo);
    Ah[r][c] = __builtin_bit_cast(__bf16, (unsigned short)(hb >> 16));
    Al[r][c] = __builtin_bit_cast(__bf16, (unsigned short)(lb >> 16));
  }
  __syncthreads();

  const int ntile = blockIdx.x * 8 + wave;
  const int n0    = ntile * 16;
  if (n0 >= V_) return;

  const int half = lane >> 4;
  const int l16  = lane & 15;
  const __bf16* bhp = Bh + (size_t)(n0 + l16) * KD + (half << 3);
  const __bf16* blp = Bl + (size_t)(n0 + l16) * KD + (half << 3);

  v8f acc[MB];
#pragma unroll
  for (int mt = 0; mt < MB; ++mt) acc[mt] = (v8f){};

  for (int k = 0; k < KD; k += 32) {
    v8bf bh0 = *(const v8bf*)(bhp + k);
    v8bf bh1 = *(const v8bf*)(bhp + k + 16);
    v8bf bl0 = *(const v8bf*)(blp + k);
    v8bf bl1 = *(const v8bf*)(blp + k + 16);
    v16bf bhf = __builtin_shufflevector(bh0, bh1, 0,1,2,3,4,5,6,7,8,9,10,11,12,13,14,15);
    v16bf blf = __builtin_shufflevector(bl0, bl1, 0,1,2,3,4,5,6,7,8,9,10,11,12,13,14,15);
#pragma unroll
    for (int mt = 0; mt < MB; ++mt) {
      const __bf16* ahp = &Ah[mt * 16 + l16][half << 3];
      const __bf16* alp = &Al[mt * 16 + l16][half << 3];
      v8bf ah0 = *(const v8bf*)(ahp + k);
      v8bf ah1 = *(const v8bf*)(ahp + k + 16);
      v8bf al0 = *(const v8bf*)(alp + k);
      v8bf al1 = *(const v8bf*)(alp + k + 16);
      v16bf ahf = __builtin_shufflevector(ah0, ah1, 0,1,2,3,4,5,6,7,8,9,10,11,12,13,14,15);
      v16bf alf = __builtin_shufflevector(al0, al1, 0,1,2,3,4,5,6,7,8,9,10,11,12,13,14,15);
      acc[mt] = __builtin_amdgcn_wmma_f32_16x16x32_bf16(false, ahf, false, bhf, (short)0, acc[mt], false, false);
      acc[mt] = __builtin_amdgcn_wmma_f32_16x16x32_bf16(false, ahf, false, blf, (short)0, acc[mt], false, false);
      acc[mt] = __builtin_amdgcn_wmma_f32_16x16x32_bf16(false, alf, false, bhf, (short)0, acc[mt], false, false);
    }
  }

  const int n = n0 + l16;
  float bv = bias[n];
#pragma unroll
  for (int mt = 0; mt < MB; ++mt) {
#pragma unroll
    for (int r = 0; r < 8; ++r) {
      int m = m0 + mt * 16 + r + (half << 3);
      __builtin_nontemporal_store(acc[mt][r] + bv, &C[(size_t)m * (size_t)V_ + n]);
    }
  }
}

// ---------------------------------------------------------------------------
// Fused LSTM step body (one 8-wave block = 8 of the 64 (mtile,htile) tiles).
// gates = xg + hprev @ Whh^T; then pointwise c/h update. Each wave owns its
// (b,h) 16x16 tile of c, so the update is local.
// ---------------------------------------------------------------------------
__device__ __forceinline__ void lstm_block(float (*S)[LDSP], int brel,
                                           const float* __restrict__ xg,
                                           const float* __restrict__ hprev,
                                           const float* __restrict__ Whh,
                                           float* __restrict__ cst,
                                           float* __restrict__ hout) {
  const int tid  = threadIdx.x;
  const int lane = tid & 31;
  const int wave = tid >> 5;
  const int tile  = brel * 8 + wave;     // 0..63
  const int mtile = brel >> 2;           // all 8 waves of a block share mtile
  const int m0    = mtile * 16;
  const int h0    = (tile & 31) * 16;

  for (int i = tid; i < 16 * H_; i += 256) {
    int r = i >> 9;
    int c = i & (H_ - 1);
    S[r][c] = hprev[(m0 + r) * H_ + c];
  }
  __syncthreads();

  const int half = lane >> 4;
  const int l16  = lane & 15;
  const float* ap  = &S[l16][half << 1];
  const float* bp0 = Whh + (size_t)(0 * H_ + h0 + l16) * H_ + (half << 1);
  const float* bp1 = Whh + (size_t)(1 * H_ + h0 + l16) * H_ + (half << 1);
  const float* bp2 = Whh + (size_t)(2 * H_ + h0 + l16) * H_ + (half << 1);
  const float* bp3 = Whh + (size_t)(3 * H_ + h0 + l16) * H_ + (half << 1);

  v8f ai = {}, af = {}, ag = {}, ao = {};
#pragma unroll 2
  for (int k = 0; k < H_; k += 4) {
    v2f hfrag = *(const v2f*)(ap + k);
    v2f bi = *(const v2f*)(bp0 + k);
    v2f bf = *(const v2f*)(bp1 + k);
    v2f bg = *(const v2f*)(bp2 + k);
    v2f bo = *(const v2f*)(bp3 + k);
    ai = __builtin_amdgcn_wmma_f32_16x16x4_f32(false, hfrag, false, bi, (short)0, ai, false, false);
    af = __builtin_amdgcn_wmma_f32_16x16x4_f32(false, hfrag, false, bf, (short)0, af, false, false);
    ag = __builtin_amdgcn_wmma_f32_16x16x4_f32(false, hfrag, false, bg, (short)0, ag, false, false);
    ao = __builtin_amdgcn_wmma_f32_16x16x4_f32(false, hfrag, false, bo, (short)0, ao, false, false);
  }

  const int hcol = h0 + l16;
#pragma unroll
  for (int r = 0; r < 8; ++r) {
    int b = m0 + r + (half << 3);
    const float* xr = xg + (size_t)b * G4H + hcol;
    float gi = ai[r] + xr[0 * H_];
    float gf = af[r] + xr[1 * H_];
    float gg = ag[r] + xr[2 * H_];
    float go = ao[r] + xr[3 * H_];
    float co = cst[(size_t)b * H_ + hcol];
    float cn = sigm(gf) * co + sigm(gi) * tanhf(gg);
    float hn = sigm(go) * tanhf(cn);
    cst[(size_t)b * H_ + hcol]  = cn;
    hout[(size_t)b * H_ + hcol] = hn;
  }
}

// ---------------------------------------------------------------------------
// One pipelined time-index launch (48 blocks x 256):
//  blocks  0..7  : layer-0 fused step t            (reads Xp[t], y0[t-1]; writes y0[t], c0)
//  blocks  8..39 : xp1[t-1] = y0[t-1] @ Wih1^T + b (256 tiles; off critical path)
//  blocks 40..47 : layer-1 fused step t-2          (reads xp1 ring, y1[t-3]; writes y1[t-2], c1)
// All per-launch data hazards are disjoint (ring-parity argument).
// ---------------------------------------------------------------------------
__global__ void megastep(
    int do0, const float* __restrict__ xg0, const float* __restrict__ h0p,
    const float* __restrict__ Whh0, float* __restrict__ c0, float* __restrict__ y0o,
    int dox, const float* __restrict__ x1in, const float* __restrict__ Wih1,
    const float* __restrict__ b1s, float* __restrict__ xp1o,
    int do2, const float* __restrict__ xg1, const float* __restrict__ h1p,
    const float* __restrict__ Whh1, float* __restrict__ c1, float* __restrict__ y1o) {
  __shared__ float S[16][LDSP];
  const int bx   = blockIdx.x;
  const int tid  = threadIdx.x;
  const int lane = tid & 31;
  const int wave = tid >> 5;

  if (bx < 8) {
    if (!do0) return;
    lstm_block(S, bx, xg0, h0p, Whh0, c0, y0o);
  } else if (bx < 40) {
    if (!dox) return;
    // input projection for layer-1 step t-1: 256 tiles = 2 mtiles x 128 ntiles
    const int tile  = (bx - 8) * 8 + wave;
    const int mtile = tile >> 7;           // blocks homogeneous in mtile (8 | 128)
    const int m0    = mtile * 16;
    const int n0    = (tile & 127) * 16;
    for (int i = tid; i < 16 * KD; i += 256) {
      int r = i >> 9;
      int c = i & (KD - 1);
      S[r][c] = x1in[(m0 + r) * KD + c];
    }
    __syncthreads();
    const int half = lane >> 4;
    const int l16  = lane & 15;
    const float* ap = &S[l16][half << 1];
    const float* bp = Wih1 + (size_t)(n0 + l16) * KD + (half << 1);
    v8f acc = {};
#pragma unroll 4
    for (int k = 0; k < KD; k += 4) {
      v2f af = *(const v2f*)(ap + k);
      v2f bf = *(const v2f*)(bp + k);
      acc = __builtin_amdgcn_wmma_f32_16x16x4_f32(false, af, false, bf,
                                                  (short)0, acc, false, false);
    }
    const int n = n0 + l16;
    float bv = b1s[n];
#pragma unroll
    for (int r = 0; r < 8; ++r) {
      int m = m0 + r + (half << 3);
      xp1o[(size_t)m * G4H + n] = acc[r] + bv;
    }
  } else {
    if (!do2) return;
    lstm_block(S, bx - 40, xg1, h1p, Whh1, c1, y1o);
  }
}

// c0[b][h] = emb[word[b]] . w2h_W[h] + w2h_b[h] -> both layers' c state
// (living in the c_final output slots). Also zeros h_{-1} and precomputes
// b1sum = bih1 + bhh1.
__global__ void init_kernel(const int* __restrict__ word, const float* __restrict__ emb,
                            const float* __restrict__ w2hW, const float* __restrict__ w2hb,
                            const float* __restrict__ bih1, const float* __restrict__ bhh1,
                            float* __restrict__ c0a, float* __restrict__ c0b,
                            float* __restrict__ zb, float* __restrict__ b1s) {
  int idx = blockIdx.x * blockDim.x + threadIdx.x;
  if (idx >= B_ * H_) return;
  if (idx < G4H) b1s[idx] = bih1[idx] + bhh1[idx];
  int b = idx >> 9;
  int h = idx & (H_ - 1);
  const float4* er = (const float4*)(emb + (size_t)word[b] * E_);
  const float4* wr = (const float4*)(w2hW + (size_t)h * E_);
  float s = 0.0f;
#pragma unroll 4
  for (int k = 0; k < E_ / 4; ++k) {
    float4 e = er[k], w = wr[k];
    s += e.x * w.x + e.y * w.y + e.z * w.z + e.w * w.w;
  }
  s += w2hb[h];
  c0a[idx] = s;
  c0b[idx] = s;
  zb[idx]  = 0.0f;
}

__global__ void hfinal_copy(const float* __restrict__ h0last, const float* __restrict__ h1last,
                            float* __restrict__ hfin) {
  int idx = blockIdx.x * blockDim.x + threadIdx.x;
  int n = B_ * H_;
  if (idx < n)          hfin[idx] = h0last[idx];
  else if (idx < 2 * n) hfin[idx] = h1last[idx - n];
}

extern "C" void kernel_launch(void* const* d_in, const int* in_sizes, int n_in,
                              void* d_out, int out_size, void* d_ws, size_t ws_size,
                              hipStream_t stream) {
  const int*   word = (const int*)  d_in[0];
  const int*   seq  = (const int*)  d_in[1];
  const float* emb  = (const float*)d_in[2];
  const float* w2hW = (const float*)d_in[3];
  const float* w2hb = (const float*)d_in[4];
  const float* Wih0 = (const float*)d_in[5];
  const float* Whh0 = (const float*)d_in[6];
  const float* bih0 = (const float*)d_in[7];
  const float* bhh0 = (const float*)d_in[8];
  const float* Wih1 = (const float*)d_in[9];
  const float* Whh1 = (const float*)d_in[10];
  const float* bih1 = (const float*)d_in[11];
  const float* bhh1 = (const float*)d_in[12];
  const float* decW = (const float*)d_in[13];
  const float* decb = (const float*)d_in[14];

  float* out = (float*)d_out;
  const size_t DEC = (size_t)TB * V_;           // 204,800,000 floats
  float* hfin = out + DEC;                      // [L,B,H]
  float* cfin = out + DEC + 2ull * B_ * H_;     // [L,B,H]
  float* c0l  = cfin;                           // layer0 c state == c_final[0]
  float* c1l  = cfin + (size_t)B_ * H_;         // layer1 c state == c_final[1]

  // workspace layout (bytes):
  //   Xp 33,554,432 | y0 8,388,608 | y1 8,388,608 | zb 65,536 | b1s 8,192 |
  //   xp1 ring 524,288 | decWh 51,200,000 | decWl 51,200,000  -> 153,329,664
  char* wsb = (char*)d_ws;
  float*  Xp    = (float*)(wsb);
  float*  y0    = (float*)(wsb + 33554432);
  float*  y1    = (float*)(wsb + 41943040);
  float*  zb    = (float*)(wsb + 50331648);
  float*  b1s   = (float*)(wsb + 50397184);
  float*  xp1   = (float*)(wsb + 50405376);
  __bf16* decWh = (__bf16*)(wsb + 50929664);
  __bf16* decWl = (__bf16*)(wsb + 102129664);
  const size_t NEED_BF16 = 153329664ull;
  const bool use_bf16 = (ws_size >= NEED_BF16);

  init_kernel<<<(B_ * H_ + 255) / 256, 256, 0, stream>>>(
      word, emb, w2hW, w2hb, bih1, bhh1, c0l, c1l, zb, b1s);

  if (use_bf16) {
    const unsigned nW = (unsigned)V_ * (unsigned)H_;   // 25,600,000
    split_bf16<<<(nW + 255u) / 256u, 256, 0, stream>>>(decW, decWh, decWl, nW);
  }

  // Layer-0 input projection for all timesteps (embedding gather in A load):
  gemm_wmma_k512<false><<<dim3(G4H / 16 / 8, TB / 16), 256, 0, stream>>>(
      emb, seq, Wih0, bih0, bhh0, Xp, G4H, G4H);

  // Pipelined recurrence: 130 launches cover layer0 t=0..127, xproj1 s=0..127,
  // layer1 s=0..127 (stage lag 0 / 1 / 2).
  const size_t BH = (size_t)B_ * H_;
  for (int t = 0; t <= T_ + 1; ++t) {
    const int do0 = (t < T_) ? 1 : 0;
    const int dox = (t >= 1 && t <= T_) ? 1 : 0;
    const int do2 = (t >= 2) ? 1 : 0;
    const int t0 = do0 ? t : (T_ - 1);          // clamped (unused when inactive)
    const int sx = dox ? (t - 1) : 0;
    const int s2 = do2 ? (t - 2) : 0;
    const float* xg0 = Xp + (size_t)t0 * B_ * G4H;
    const float* h0p = (t0 == 0) ? zb : (y0 + (size_t)(t0 - 1) * BH);
    float*       y0o = y0 + (size_t)t0 * BH;
    const float* x1in = y0 + (size_t)sx * BH;
    float*       xp1o = xp1 + (size_t)(sx & 1) * B_ * G4H;
    const float* xg1 = xp1 + (size_t)(s2 & 1) * B_ * G4H;
    const float* h1p = (s2 == 0) ? zb : (y1 + (size_t)(s2 - 1) * BH);
    float*       y1o = y1 + (size_t)s2 * BH;
    megastep<<<48, 256, 0, stream>>>(do0, xg0, h0p, Whh0, c0l, y0o,
                                     dox, x1in, Wih1, b1s, xp1o,
                                     do2, xg1, h1p, Whh1, c1l, y1o);
  }

  // Decoder: decoded[T*B, V] = y1 @ dec_W^T + dec_b
  const int ntiles = V_ / 16;  // 3125
  if (use_bf16) {
    gemm_dec_bf16split<<<dim3((ntiles + 7) / 8, TB / (16 * MB)), 256, 0, stream>>>(
        y1, decWh, decWl, decb, out);
  } else {
    gemm_wmma_k512<true><<<dim3((ntiles + 7) / 8, TB / 16), 256, 0, stream>>>(
        y1, nullptr, decW, decb, nullptr, out, V_, V_);
  }

  hfinal_copy<<<(2 * B_ * H_ + 255) / 256, 256, 0, stream>>>(
      y0 + (size_t)(T_ - 1) * BH, y1 + (size_t)(T_ - 1) * BH, hfin);
}
